// NNMF2d_23046794510685
// MI455X (gfx1250) — matrix operand order repaired and checked by reading, not compiled, not run on hardware
//
#include <hip/hip_runtime.h>

// -------------------------------------------------------------------------
// NNMF2d on MI455X (gfx1250, wave32, WMMA bf16 16x16x32, f32 accumulate)
//
// Positions (B*H*W = 8192) are independent: each workgroup owns 32 positions
// (2 row tiles of 16) and runs all 20 multiplicative-update iterations
// locally.  Per iteration:
//   phase1: rec = h @ w        (A = h bf16 in LDS, B = packed wB1, f32 acc)
//           q   = x/(rec+eps)  (x f32 in LDS)      -> abuf_q (bf16)
//   phase2: t   = q @ w^T      (A = q bf16 in LDS, B = packed wB2)
//           h  *= t ; h /= rowsum(h)   (register C/D fragments + ds_add_f32)
//
// Round-3 changes (roofline-driven): each wave reuses every B fragment for
// TWO row tiles -> L2 w-traffic per FLOP halves (~10.5 GB -> ~5.2 GB total),
// and the inner loop issues 2 WMMAs per 32B global B load. x moved from
// registers to LDS to stay under the VGPR spill cliff found in round 1.
// LDS (~130 KB/WG) is dynamic — CDNA5 WGs may use up to 320 KB.
// -------------------------------------------------------------------------

typedef __attribute__((ext_vector_type(16))) __bf16 v16bf;
typedef __attribute__((ext_vector_type(8)))  float  v8f;

#define CIN   512
#define COUT  512
#define ITERS 20
#define MPOS  32    // positions per workgroup (2 row tiles)
#define ABST  520   // padded LDS row stride -> conflict-free b128 A loads

// dynamic-LDS carve-out (bytes)
#define OFF_ABUF_H  0
#define OFF_ABUF_Q  (MPOS * ABST * 2)                    // 33280
#define OFF_XS      (OFF_ABUF_Q + MPOS * ABST * 2)       // 66560
#define OFF_SUMS    (OFF_XS + MPOS * ABST * 4)           // 133120
#define SMEM_BYTES  (OFF_SUMS + MPOS * 4)                // 133248

union AFrag { uint4 u[2]; v16bf v; };

// 16-bit A-matrix 16x32 in this code's k order (not the ISA table's): lane L holds row M=L%16;
// half = L/16 selects K = k0+16h..+15 (elems 0-15, contiguous, matches packed B)
__device__ __forceinline__ v16bf load_a_frag(const __bf16* row, int k0, int hf) {
    AFrag f;
    f.u[0] = *(const uint4*)(row + k0 + 16 * hf);
    f.u[1] = *(const uint4*)(row + k0 + 16 * hf + 8);
    return f.v;
}

// 16-step GEMM panel over TWO row tiles sharing each B fragment.
// B software-pipelined 1 deep (2 fragments live).
__device__ __forceinline__ void gemm_panel2(const __bf16* arow0, const __bf16* arow1,
                                            int hf, const v16bf* __restrict__ bp,
                                            v8f& acc0, v8f& acc1) {
    acc0 = (v8f){0.f, 0.f, 0.f, 0.f, 0.f, 0.f, 0.f, 0.f};
    acc1 = (v8f){0.f, 0.f, 0.f, 0.f, 0.f, 0.f, 0.f, 0.f};
    v16bf b = bp[0];
    #pragma unroll 1
    for (int kt = 0; kt < 15; ++kt) {
        v16bf bn = bp[(kt + 1) * 32];                    // prefetch next B (L2)
        v16bf a0 = load_a_frag(arow0, kt * 32, hf);
        acc0 = __builtin_amdgcn_wmma_f32_16x16x32_bf16(false, a0, false, b,
                                                       (short)0, acc0, false, false);
        v16bf a1 = load_a_frag(arow1, kt * 32, hf);
        acc1 = __builtin_amdgcn_wmma_f32_16x16x32_bf16(false, a1, false, b,
                                                       (short)0, acc1, false, false);
        b = bn;
    }
    v16bf a0 = load_a_frag(arow0, 15 * 32, hf);
    acc0 = __builtin_amdgcn_wmma_f32_16x16x32_bf16(false, a0, false, b,
                                                   (short)0, acc0, false, false);
    v16bf a1 = load_a_frag(arow1, 15 * 32, hf);
    acc1 = __builtin_amdgcn_wmma_f32_16x16x32_bf16(false, a1, false, b,
                                                   (short)0, acc1, false, false);
}

// ---------------- prep 1: rowsum[o] = sum_c |w[o][c]| -----------------------
__global__ void __launch_bounds__(256) nnmf_rowsum(const float* __restrict__ w,
                                                   float* __restrict__ rowsum) {
    const int o = blockIdx.x, tid = threadIdx.x;
    float s = fabsf(w[o * CIN + tid]) + fabsf(w[o * CIN + tid + 256]);
    #pragma unroll
    for (int m = 1; m < 32; m <<= 1) s += __shfl_xor(s, m, 32);
    __shared__ float ws8[8];
    if ((tid & 31) == 0) ws8[tid >> 5] = s;
    __syncthreads();
    if (tid == 0) {
        float t = 0.f;
        #pragma unroll
        for (int i = 0; i < 8; ++i) t += ws8[i];
        rowsum[o] = t;
    }
}

// ---------------- prep 2: pack normalized w into B-fragment layouts ---------
// B-frag (32x16, bf16): lane = halfB*16 + n holds column N=n, K = kt*32+16*halfB+e
// wB1: K=o (Cout), N=c (Cin).   wB2: K=c (Cin), N=o (Cout)  (i.e. w^T)
__global__ void __launch_bounds__(256) nnmf_pack(const float* __restrict__ w,
                                                 const float* __restrict__ rowsum,
                                                 __bf16* __restrict__ wB1,
                                                 __bf16* __restrict__ wB2) {
    const int idx = blockIdx.x * 256 + threadIdx.x;   // 0 .. 512*512-1
    const int o = idx >> 9, c = idx & 511;
    const float v = fabsf(w[idx]) * __builtin_amdgcn_rcpf(rowsum[o] + 1e-19f);
    const __bf16 bv = (__bf16)v;
    {   // GEMM1 operand: K=o, N=c
        const int nt = c >> 4, kt = o >> 5, r = o & 31;
        wB1[((((nt * 16 + kt) * 32) + (r >> 4) * 16 + (c & 15)) << 4) + (r & 15)] = bv;
    }
    {   // GEMM2 operand: K=c, N=o
        const int nt = o >> 4, kt = c >> 5, r = c & 31;
        wB2[((((nt * 16 + kt) * 32) + (r >> 4) * 16 + (o & 15)) << 4) + (r & 15)] = bv;
    }
}

// ---------------- main persistent kernel ------------------------------------
__global__ void __launch_bounds__(256) nnmf_iterate(const float* __restrict__ x,
                                                    const v16bf* __restrict__ wB1,
                                                    const v16bf* __restrict__ wB2,
                                                    float* __restrict__ out) {
    extern __shared__ __align__(16) char smem[];
    __bf16* abuf_h = (__bf16*)(smem + OFF_ABUF_H);   // bf16 h     (A of GEMM1)
    __bf16* abuf_q = (__bf16*)(smem + OFF_ABUF_Q);   // bf16 x/rec (A of GEMM2)
    float*  x_s    = (float*)(smem + OFF_XS);        // f32 x tile
    float*  sums   = (float*)(smem + OFF_SUMS);      // per-position row sums

    const int tid  = threadIdx.x;
    const int wv   = tid >> 5;       // wave 0..7 -> owns col tiles wv+8t
    const int lane = tid & 31;
    const int n    = lane & 15;      // C/D col within tile; A row index
    const int hf   = lane >> 4;      // half-wave select

    float h_frag[2][4][8];           // h, C/D layout: m = r*16+g+8*hf, chan=(wv+8t)*16+n

    // ---- cooperative x tile load: x_s[m][c] = x[b, c, h, w]
    for (int i = tid; i < MPOS * CIN; i += 256) {
        const int m  = i >> 9;
        const int c  = i & 511;
        const int p  = blockIdx.x * MPOS + m;
        const int b  = p >> 10;
        const int hw = p & 1023;
        x_s[m * ABST + c] = x[(((b << 9) + c) << 10) + hw];
    }
    // ---- init h fragments and bf16 h operand
    #pragma unroll
    for (int r = 0; r < 2; ++r)
        #pragma unroll
        for (int t = 0; t < 4; ++t)
            #pragma unroll
            for (int g = 0; g < 8; ++g) h_frag[r][t][g] = 1.0f / (float)COUT;
    const __bf16 hinit = (__bf16)(1.0f / (float)COUT);
    for (int i = tid; i < MPOS * COUT; i += 256)
        abuf_h[(i >> 9) * ABST + (i & 511)] = hinit;

    const __bf16* arow_h0 = abuf_h + n * ABST;
    const __bf16* arow_h1 = abuf_h + (16 + n) * ABST;
    const __bf16* arow_q0 = abuf_q + n * ABST;
    const __bf16* arow_q1 = abuf_q + (16 + n) * ABST;

    for (int it = 0; it < ITERS; ++it) {
        __syncthreads();                       // abuf_h/x_s ready; prev sums reads sealed
        if (tid < MPOS) sums[tid] = 0.0f;

        // ---- phase 1: rec = h @ w ; q = x/(rec+eps) -> abuf_q (bf16)
        #pragma unroll
        for (int t = 0; t < 4; ++t) {
            const int nt = wv + 8 * t;
            __builtin_prefetch(wB2 + nt * 512 + lane, 0, 3);  // warm phase-2 stream
            v8f acc0, acc1;
            gemm_panel2(arow_h0, arow_h1, hf, wB1 + nt * 512 + lane, acc0, acc1);
            const int c0 = nt << 4;
            #pragma unroll
            for (int g = 0; g < 8; ++g) {
                const int m0 = g + 8 * hf;
                const float q0 = x_s[m0 * ABST + c0 + n] *
                                 __builtin_amdgcn_rcpf(acc0[g] + 1e-20f);
                abuf_q[m0 * ABST + c0 + n] = (__bf16)q0;
                const int m1 = m0 + 16;
                const float q1 = x_s[m1 * ABST + c0 + n] *
                                 __builtin_amdgcn_rcpf(acc1[g] + 1e-20f);
                abuf_q[m1 * ABST + c0 + n] = (__bf16)q1;
            }
        }
        __syncthreads();                       // abuf_q complete; sums zeroed

        // ---- phase 2: t = q @ w^T ; h *= t ; accumulate row sums
        #pragma unroll
        for (int t = 0; t < 4; ++t) {
            const int nt = wv + 8 * t;
            v8f acc0, acc1;
            gemm_panel2(arow_q0, arow_q1, hf, wB2 + nt * 512 + lane, acc0, acc1);
            #pragma unroll
            for (int g = 0; g < 8; ++g) {
                {
                    const float hv = h_frag[0][t][g] * acc0[g];
                    h_frag[0][t][g] = hv;
                    float rr = hv;             // reduce over 16 lanes of this half
                    rr += __shfl_xor(rr, 1, 32);
                    rr += __shfl_xor(rr, 2, 32);
                    rr += __shfl_xor(rr, 4, 32);
                    rr += __shfl_xor(rr, 8, 32);
                    if (n == 0) atomicAdd(&sums[g + 8 * hf], rr);        // ds_add_f32
                }
                {
                    const float hv = h_frag[1][t][g] * acc1[g];
                    h_frag[1][t][g] = hv;
                    float rr = hv;
                    rr += __shfl_xor(rr, 1, 32);
                    rr += __shfl_xor(rr, 2, 32);
                    rr += __shfl_xor(rr, 4, 32);
                    rr += __shfl_xor(rr, 8, 32);
                    if (n == 0) atomicAdd(&sums[16 + g + 8 * hf], rr);
                }
            }
        }
        __syncthreads();                       // sums complete

        // ---- normalize h, refresh bf16 h operand
        float inv[2][8];
        #pragma unroll
        for (int r = 0; r < 2; ++r)
            #pragma unroll
            for (int g = 0; g < 8; ++g)
                inv[r][g] = __builtin_amdgcn_rcpf(sums[r * 16 + g + 8 * hf] + 1e-19f);
        #pragma unroll
        for (int t = 0; t < 4; ++t) {
            const int c0 = (wv + 8 * t) << 4;
            #pragma unroll
            for (int r = 0; r < 2; ++r)
                #pragma unroll
                for (int g = 0; g < 8; ++g) {
                    const float hv = h_frag[r][t][g] * inv[r][g];
                    h_frag[r][t][g] = hv;
                    abuf_h[(r * 16 + g + 8 * hf) * ABST + c0 + n] = (__bf16)hv;
                }
        }
    }

    // ---- write out[b, o, h, w]
    #pragma unroll
    for (int t = 0; t < 4; ++t) {
        const int o0 = (wv + 8 * t) << 4;
        #pragma unroll
        for (int r = 0; r < 2; ++r)
            #pragma unroll
            for (int g = 0; g < 8; ++g) {
                const int m  = r * 16 + g + 8 * hf;
                const int p  = blockIdx.x * MPOS + m;
                const int b  = p >> 10;
                const int hw = p & 1023;
                out[(((b << 9) + (o0 + n)) << 10) + hw] = h_frag[r][t][g];
            }
    }
}

// ---------------------------------------------------------------------------
extern "C" void kernel_launch(void* const* d_in, const int* in_sizes, int n_in,
                              void* d_out, int out_size, void* d_ws, size_t ws_size,
                              hipStream_t stream) {
    (void)in_sizes; (void)n_in; (void)out_size; (void)ws_size;
    const float* x = (const float*)d_in[0];     // [8, 512, 32, 32] f32
    const float* w = (const float*)d_in[1];     // [512, 512] f32

    // workspace: rowsum (2KB) | wB1 packed bf16 (512KB) | wB2 packed bf16 (512KB)
    char*   ws      = (char*)d_ws;
    float*  rowsum  = (float*)ws;
    __bf16* wB1     = (__bf16*)(ws + 4096);
    __bf16* wB2     = (__bf16*)(ws + 4096 + 512 * 1024);

    nnmf_rowsum <<<COUT, 256, 0, stream>>>(w, rowsum);
    nnmf_pack   <<<(COUT * CIN) / 256, 256, 0, stream>>>(w, rowsum, wB1, wB2);
    nnmf_iterate<<<8192 / MPOS, 256, SMEM_BYTES, stream>>>(x, (const v16bf*)wB1,
                                                           (const v16bf*)wB2,
                                                           (float*)d_out);
}
